// WeightedAngularAEVComputer_84335977825046
// MI455X (gfx1250) — matrix operand order repaired and checked by reading, not compiled, not run on hardware
//
#include <hip/hip_runtime.h>

typedef __attribute__((ext_vector_type(16))) _Float16 v16h;
typedef __attribute__((ext_vector_type(8)))  _Float16 v8h;
typedef __attribute__((ext_vector_type(8)))  float    v8f;

#define BATCH  16
#define NATOM  64
#define NTRIP  1953   // C(63,2) triplets per center atom
#define NCHUNK 62     // ceil(1953 / 32) K-chunks per center

// GA[b,i,m+8n] = sum_t base_t * f1_t[m] * f2_t[n]
//  -> per (b,i): 8x4 GEMM, K = NTRIP, run on v_wmma_f32_16x16x32_f16
__global__ __launch_bounds__(256) void aev_angular_wmma(
    const float* __restrict__ dmat,   // [B,64,64]
    const float* __restrict__ znum,   // [B,64]
    const int*   __restrict__ trip,   // [64,1953,3]
    float*       __restrict__ out)    // [B,64,32]
{
  __shared__ __attribute__((aligned(16))) float s_d[NATOM * NATOM]; // 16 KB
  __shared__ float s_z[NATOM];
  // per-wave transposed staging, padded to 16 rows so fragment gathers are
  // branchless: rows 8..15 (A) / 4..15 (B) are zeroed once and never written.
  __shared__ __attribute__((aligned(32))) _Float16 s_a1[8][16][32]; // 8 KB
  __shared__ __attribute__((aligned(32))) _Float16 s_f2[8][16][32]; // 8 KB

  const int b    = blockIdx.x;
  const int tid  = threadIdx.x;
  const int w    = tid >> 5;    // wave id (wave32)
  const int lane = tid & 31;

  // stage this batch's 64x64 distance matrix + atomic numbers in LDS
  for (int idx = tid; idx < NATOM * NATOM; idx += 256)
    s_d[idx] = dmat[b * NATOM * NATOM + idx];
  if (tid < NATOM) s_z[tid] = znum[b * NATOM + tid];
  __syncthreads();

  // zero the padding rows of this wave's staging area (written once; the
  // hot loop only ever writes rows 0..7 / 0..3). Same-wave LDS is in-order.
  #pragma unroll
  for (int m = 8; m < 16; ++m) s_a1[w][m][lane] = (_Float16)0.0f;
  #pragma unroll
  for (int n = 4; n < 16; ++n) s_f2[w][n][lane] = (_Float16)0.0f;

  // cos/sin of ShfZ = (2m+1)*pi/16, and ShfA
  const float CZ[8] = { 0.98078528f,  0.83146961f,  0.55557023f,  0.19509032f,
                       -0.19509032f, -0.55557023f, -0.83146961f, -0.98078528f};
  const float SZ[8] = { 0.19509032f,  0.55557023f,  0.83146961f,  0.98078528f,
                        0.98078528f,  0.83146961f,  0.55557023f,  0.19509032f};
  const float SA[4] = {0.9f, 1.55f, 2.2f, 2.85f};

  const int mrow   = lane & 15;               // A row / B col owned by this lane
  const int kbaseA = (lane < 16) ? 0 : 8;     // 16-bit A 16x32 lane->K mapping
  const int kbaseB = (lane < 16) ? 0 : 8;     // 16-bit B 32x16 lane->K mapping

  for (int iq = 0; iq < 8; ++iq) {
    const int i = w + 8 * iq;                 // center atom for this wave
    const int* __restrict__ tpi = trip + (long)i * NTRIP * 3;

    v8f acc;
    #pragma unroll
    for (int e = 0; e < 8; ++e) acc[e] = 0.0f;

    for (int c = 0; c < NCHUNK; ++c) {
      const int  t     = c * 32 + lane;
      const bool valid = (t < NTRIP);
      const int  tt    = valid ? t : (NTRIP - 1);
      const int  j     = tpi[tt * 3 + 1];
      const int  k     = tpi[tt * 3 + 2];

      // prefetch next chunk's triplet record into near cache
      const int tn = (t + 32 < NTRIP) ? (t + 32) : (NTRIP - 1);
      __builtin_prefetch(tpi + tn * 3, 0, 3);

      const float dij = s_d[i * NATOM + j];
      const float dik = s_d[i * NATOM + k];
      const float djk = s_d[j * NATOM + k];
      const float zj  = s_z[j];               // unconditional: keeps the
      const float zk  = s_z[k];               // mask branchless (cndmask)

      const bool mask = valid &&
                        (dij < 3.5f) && (dik < 3.5f) && (djk < 1e20f) &&
                        (dij != 0.0f) && (dik != 0.0f) && (djk != 0.0f);

      const float denom = fmaxf(2.0f * dij * dik, 1e-10f);
      float ca = 0.95f * (dij * dij + dik * dik - djk * djk) / denom; // cos(alpha)
      ca = fminf(fmaxf(ca, -1.0f), 1.0f);
      const float sa = sqrtf(fmaxf(1.0f - ca * ca, 0.0f));            // sin(alpha)

      const float fcij = __cosf(0.8975979010f * dij) * 0.5f + 0.5f;   // f_C(d_ij)
      const float fcik = __cosf(0.8975979010f * dik) * 0.5f + 0.5f;   // f_C(d_ik)
      float base = zj * zk * 2.0f * fcij * fcik;
      base = mask ? base : 0.0f;              // single v_cndmask, no branch
      const float savg = 0.5f * (dij + dik);

      // A[m, t] = base * ((1 + cos(alpha - ShfZ[m]))/2)^32 (5 squarings)
      #pragma unroll
      for (int m = 0; m < 8; ++m) {
        const float y  = 0.5f * (1.0f + ca * CZ[m] + sa * SZ[m]);
        const float y2 = y * y, y4 = y2 * y2, y8 = y4 * y4, y16 = y8 * y8;
        s_a1[w][m][lane] = (_Float16)(base * (y16 * y16));
      }
      // B[t, n] = exp(-8 * ((d_ij+d_ik)/2 - ShfA[n])^2)
      #pragma unroll
      for (int n = 0; n < 4; ++n) {
        const float dd = savg - SA[n];
        s_f2[w][n][lane] = (_Float16)__expf(-8.0f * dd * dd);
      }

      // make the cross-lane staging visible before the fragment gathers
      asm volatile("s_wait_dscnt 0" ::: "memory");

      // branchless fragment gathers (padding rows supply the zeros)
      // A: elems 0..7 -> K = kbaseA+0..7, elems 8..15 -> K = 16+kbaseA+0..7
      const v8h alo = *(const v8h*)&s_a1[w][mrow][kbaseA];
      const v8h ahi = *(const v8h*)&s_a1[w][mrow][kbaseA + 16];
      // B: elems 0..7 -> K = kbaseB+0..7, elems 8..15 -> K = 16+kbaseB+0..7
      const v8h blo = *(const v8h*)&s_f2[w][mrow][kbaseB];
      const v8h bhi = *(const v8h*)&s_f2[w][mrow][kbaseB + 16];
      v16h Af, Bf;
      #pragma unroll
      for (int e = 0; e < 8; ++e) {
        Af[e] = alo[e]; Af[e + 8] = ahi[e];
        Bf[e] = blo[e]; Bf[e + 8] = bhi[e];
      }

      // D(16x16,f32) += A(16x32,f16) x B(32x16,f16): 32 triplets per issue
      acc = __builtin_amdgcn_wmma_f32_16x16x32_f16(
          false, Af, false, Bf, (short)0, acc, false, false);
    }

    // D layout: lane n (0..3) holds column n, VGPR v holds row m=v (0..7)
    if (lane < 4) {
      float* op = out + ((size_t)(b * NATOM + i)) * 32 + lane * 8;
      #pragma unroll
      for (int e = 0; e < 8; ++e) op[e] = acc[e];
    }
  }
}

extern "C" void kernel_launch(void* const* d_in, const int* in_sizes, int n_in,
                              void* d_out, int out_size, void* d_ws, size_t ws_size,
                              hipStream_t stream) {
  const float* dmat = (const float*)d_in[0];  // distance_matrices [16,64,64] f32
  const float* znum = (const float*)d_in[1];  // atomic_numbers_batch [16,64] f32
  const int*   trip = (const int*)d_in[2];    // triplets [64,1953,3] i32
  float*       out  = (float*)d_out;          // GA_batch [16,64,32] f32

  aev_angular_wmma<<<dim3(BATCH), dim3(256), 0, stream>>>(dmat, znum, trip, out);
}
